// DAMDiscreteHopfield_4243427688538
// MI455X (gfx1250) — matrix operand order repaired and checked by reading, not compiled, not run on hardware
//
#include <hip/hip_runtime.h>
#include <hip/hip_bf16.h>
#include <stdint.h>

// ---------------------------------------------------------------------------
// Problem sizes (fixed by the reference)
// ---------------------------------------------------------------------------
#define N_PATTERNS 8192
#define N_BITS     4096
#define WORDS_PER_COL (N_PATTERNS / 32)   // 256 uint32 per packed column

typedef __attribute__((ext_vector_type(16))) _Float16 v16h;
typedef __attribute__((ext_vector_type(8)))  float    v8f;
typedef __attribute__((ext_vector_type(4)))  float    f32x4;
typedef __attribute__((ext_vector_type(4)))  int      i32x4;

// ---- async global->LDS support (gfx1250), guarded for toolchain safety ----
#if defined(__has_builtin)
#if __has_builtin(__builtin_amdgcn_global_load_async_to_lds_b128) && \
    __has_builtin(__builtin_amdgcn_s_wait_asynccnt)
#define USE_ASYNC_LDS 1
#endif
#endif
#ifndef USE_ASYNC_LDS
#define USE_ASYNC_LDS 0
#endif

#if USE_ASYNC_LDS
// builtin prototype: (AS1 v4i* src, AS3 v4i* dst, imm offset, imm cpol)
typedef __attribute__((address_space(1))) i32x4* as1_i32x4p;
typedef __attribute__((address_space(3))) i32x4* as3_i32x4p;
#define WAIT_ASYNCCNT(n) __builtin_amdgcn_s_wait_asynccnt(n)
#else
#define WAIT_ASYNCCNT(n)
#endif

// ---------------------------------------------------------------------------
// Kernel 1: h0 = W @ state using v_wmma_f32_16x16x32_f16.
// Each wave computes a 16-row strip of h0. B broadcasts the state chunk to all
// 16 columns, so every column of D equals the 16-row slice of h0.
// ---------------------------------------------------------------------------
__global__ __launch_bounds__(256) void hopfield_matvec_wmma(
    const float* __restrict__ W,      // [8192, 4096] row-major, values +-1
    const float* __restrict__ state,  // [4096] values +-1
    float* __restrict__ h0)           // [8192]
{
    const int lane  = threadIdx.x & 31;
    const int wave  = threadIdx.x >> 5;
    const int m0    = (blockIdx.x * 8 + wave) * 16;   // strip base row
    const int mrow  = m0 + (lane & 15);               // A: lanes 0-15 & 16-31 both map M=0..15
    const int khalf = lane >> 4;                      // selects which K-half this lane holds
    const float* rowp = W + (size_t)mrow * N_BITS;

    v8f c = {};
    for (int kb = 0; kb < N_BITS; kb += 32) {
        // ---- A tile (16x32 f16): lane<16 holds K {0..7,16..23}, lane>=16 {8..15,24..31}
        const int baseA = kb + khalf * 8;
        f32x4 x0 = *(const f32x4*)(rowp + baseA);
        f32x4 x1 = *(const f32x4*)(rowp + baseA + 4);
        f32x4 x2 = *(const f32x4*)(rowp + baseA + 16);
        f32x4 x3 = *(const f32x4*)(rowp + baseA + 20);
        v16h a;
#pragma unroll
        for (int j = 0; j < 4; ++j) {
            a[j]      = (_Float16)x0[j];
            a[4 + j]  = (_Float16)x1[j];
            a[8 + j]  = (_Float16)x2[j];
            a[12 + j] = (_Float16)x3[j];
        }
        // ---- B tile (32x16 f16): broadcast the state chunk for this lane's K-half
        // to every column => all 16 output columns identical (= h0 slice).
        const int baseB = kb + khalf * 8;
        f32x4 s0 = *(const f32x4*)(state + baseB);
        f32x4 s1 = *(const f32x4*)(state + baseB + 4);
        f32x4 s2 = *(const f32x4*)(state + baseB + 16);
        f32x4 s3 = *(const f32x4*)(state + baseB + 20);
        v16h b;
#pragma unroll
        for (int j = 0; j < 4; ++j) {
            b[j]      = (_Float16)s0[j];
            b[4 + j]  = (_Float16)s1[j];
            b[8 + j]  = (_Float16)s2[j];
            b[12 + j] = (_Float16)s3[j];
        }
        // D = A*B + C  (f32 accumulate; +-1 products are exact)
        c = __builtin_amdgcn_wmma_f32_16x16x32_f16(
                /*neg_a=*/false, a, /*neg_b=*/false, b,
                /*c_mod=*/(short)0, c, /*reuse_a=*/false, /*reuse_b=*/false);
    }

    // C/D layout: lanes 0-15 -> M = r (vgpr r), lanes 16-31 -> M = r + 8.
    // All columns identical; let column-0 lanes (lane 0 and lane 16) write.
    if ((lane & 15) == 0) {
        const int mb = m0 + khalf * 8;
#pragma unroll
        for (int r = 0; r < 8; ++r) h0[mb + r] = c[r];
    }
}

// ---------------------------------------------------------------------------
// Kernel 2: transpose + bit-pack W. colbits[i][w] bit j = (W[32w+j, i] < 0).
// 32x traffic reduction for the serial phase (4 MB total, L2-resident).
// ---------------------------------------------------------------------------
__global__ __launch_bounds__(256) void hopfield_pack(
    const float* __restrict__ W,       // [8192, 4096]
    uint32_t* __restrict__ colbits)    // [4096][256]
{
    const int i = blockIdx.x * 256 + threadIdx.x;  // column (bit index)
    const int w = blockIdx.y;                      // 32-pattern word
    const int p0 = w * 32;
    uint32_t bits = 0u;
#pragma unroll 4
    for (int j = 0; j < 32; ++j) {
        float v = W[(size_t)(p0 + j) * N_BITS + i];   // coalesced across threads
        bits |= (v < 0.0f ? 1u : 0u) << j;
    }
    colbits[(size_t)i * WORDS_PER_COL + w] = bits;
}

// ---------------------------------------------------------------------------
// Serial greedy descent: one workgroup, 1024 threads (32 wave32 waves).
// h[8192] in registers (8/thread), vals[4096] in LDS. The next step's packed
// column is fetched with GLOBAL_LOAD_ASYNC_TO_LDS_B128 into a double buffer
// while the current step computes -> per-step L2 latency fully hidden.
// Only 2 workgroup barriers per step.
// ---------------------------------------------------------------------------
__device__ __forceinline__ float wave_reduce_add(float v) {
#pragma unroll
    for (int off = 16; off > 0; off >>= 1) v += __shfl_xor(v, off, 32);
    return v;
}

__device__ __forceinline__ void stage_col(const uint32_t* __restrict__ colbits,
                                          int i, uint32_t* dstLds, int tid) {
    if (tid < WORDS_PER_COL / 4) {   // 64 lanes x 16B = 1 KB
        const uint32_t* src = colbits + (size_t)i * WORDS_PER_COL + tid * 4;
#if USE_ASYNC_LDS
        __builtin_amdgcn_global_load_async_to_lds_b128(
            (as1_i32x4p)(const i32x4*)src,
            (as3_i32x4p)(i32x4*)(dstLds + tid * 4),
            0, 0);
#else
        *(uint4*)(dstLds + tid * 4) = *(const uint4*)src;
#endif
    }
}

__global__ __launch_bounds__(1024) void hopfield_serial(
    const uint32_t* __restrict__ colbits,  // [4096][256]
    const float* __restrict__ h0,          // [8192]
    const float* __restrict__ state,       // [4096]
    const int* __restrict__ perm,          // [4096]
    float* __restrict__ out)               // [4096]
{
    __shared__ float    vals[N_BITS];            // 16 KB
    __shared__ uint32_t col[2][WORDS_PER_COL];   // 2 KB double-buffered column
    __shared__ float    wsum[32];

    const int tid = threadIdx.x;

    // kick off the async stage of the first column immediately
    stage_col(colbits, perm[0], col[0], tid);

    // init vals from state
    for (int i = tid; i < N_BITS; i += 1024) vals[i] = state[i];

    // each thread owns 8 consecutive h entries (patterns 8*tid .. 8*tid+7)
    float h[8];
#pragma unroll
    for (int j = 0; j < 8; ++j) h[j] = h0[tid * 8 + j];

    // initial energy: prev = -sum(relu(h)^2), computed redundantly & uniformly
    float e = 0.0f;
#pragma unroll
    for (int j = 0; j < 8; ++j) {
        float r = fmaxf(h[j], 0.0f);
        e += r * r;
    }
    e = wave_reduce_add(e);
    if ((tid & 31) == 0) wsum[tid >> 5] = e;
    __syncthreads();
    float prev = -wave_reduce_add(wsum[tid & 31]);   // uniform across all threads

    for (int t = 0; t < N_BITS; ++t) {
        const int buf = t & 1;
        const int i = perm[t];                  // uniform scalar load

        // issue next column's async stage into the other buffer, then wait for
        // the OLDER transfer (in-order completion => asynccnt<=1 suffices)
        if (t + 1 < N_BITS) {
            stage_col(colbits, perm[t + 1], col[buf ^ 1], tid);
            WAIT_ASYNCCNT(1);
        } else {
            WAIT_ASYNCCNT(0);
        }
        __syncthreads();                        // barrier A: col[buf] visible

        const float vi = vals[i];               // LDS broadcast read
        const uint32_t word = col[buf][tid >> 2];
        const uint32_t bits = (word >> ((tid & 3) * 8)) & 0xFFu;

        float hn[8];
        float ep = 0.0f;
#pragma unroll
        for (int j = 0; j < 8; ++j) {
            // bit=1 means W[p,i] = -1 ;  h_new = h - 2*vi*W[p,i]
            float wji = ((bits >> j) & 1u) ? -1.0f : 1.0f;
            hn[j] = h[j] - 2.0f * vi * wji;
            float r = fmaxf(hn[j], 0.0f);
            ep += r * r;
        }
        ep = wave_reduce_add(ep);
        if ((tid & 31) == 0) wsum[tid >> 5] = ep;
        __syncthreads();                        // barrier B: wsum ready

        // every wave redundantly reduces the same 32 values in the same order
        // -> identical e_new and accept decision in every thread (uniform)
        const float eNew = -wave_reduce_add(wsum[tid & 31]);
        if (eNew < prev) {
            prev = eNew;
#pragma unroll
            for (int j = 0; j < 8; ++j) h[j] = hn[j];
            if (tid == 0) vals[i] = -vi;        // index i touched exactly once
        }
        // col[buf^1] was last read in step t-1; that step's barrier B separates its
        // rewrite from those reads -> no third barrier needed.
    }

    __syncthreads();
    for (int i = tid; i < N_BITS; i += 1024) out[i] = vals[i];
}

// ---------------------------------------------------------------------------
// Launcher
// ---------------------------------------------------------------------------
extern "C" void kernel_launch(void* const* d_in, const int* in_sizes, int n_in,
                              void* d_out, int out_size, void* d_ws, size_t ws_size,
                              hipStream_t stream) {
    const float* W     = (const float*)d_in[0];   // [8192,4096]
    const float* state = (const float*)d_in[1];   // [4096]
    const int*   perm  = (const int*)d_in[2];     // [4096]
    float*       out   = (float*)d_out;           // [4096]

    float*    h0      = (float*)d_ws;                                   // 32 KB
    uint32_t* colbits = (uint32_t*)((char*)d_ws + N_PATTERNS * sizeof(float)); // 4 MB

    // 1) h0 = W @ state  (WMMA, 512 waves)
    hopfield_matvec_wmma<<<N_PATTERNS / (8 * 16), 256, 0, stream>>>(W, state, h0);

    // 2) bit-pack transposed W (4 MB, L2-resident for the serial phase)
    hopfield_pack<<<dim3(N_BITS / 256, N_PATTERNS / 32), 256, 0, stream>>>(W, colbits);

    // 3) serial greedy descent, single workgroup, async double-buffered columns
    hopfield_serial<<<1, 1024, 0, stream>>>(colbits, h0, state, perm, out);
}
